// MultiHeadSelfAttentionLayer_24455543783434
// MI455X (gfx1250) — matrix operand order repaired and checked by reading, not compiled, not run on hardware
//
#include <hip/hip_runtime.h>

typedef __attribute__((ext_vector_type(16))) _Float16 v16h;
typedef __attribute__((ext_vector_type(8)))  float    v8f;

#define B_   2
#define N_   2048
#define D_   1024
#define H_   16
#define HD_  64
#define NT_  (B_ * N_)          // 4096 tokens
#define CHUNK 512               // keys per online-softmax chunk

union U16H { uint4 q[2]; v16h v; };

// A-operand (16x32 f16, row-major source): lane L<16 -> row m0+L, K=0..15;
// lane L+16 -> row m0+L, K=16..31. Two contiguous 16B loads per lane.
__device__ inline v16h load_a16(const _Float16* A, int lda, int m0, int k0, int lane) {
  int half = lane >> 4;
  int row  = m0 + (lane & 15);
  const _Float16* p = A + (size_t)row * lda + k0 + half * 16;
  U16H u;
  u.q[0] = *(const uint4*)(p);
  u.q[1] = *(const uint4*)(p + 8);
  return u.v;
}

// B-operand (32x16 f16) from a transposed (n-major) matrix Bt[n][k]:
// lane l holds column n = l&15, K = (l>>4)*16 + 0..15 sequential. Contiguous 32B.
__device__ inline v16h load_bt16(const _Float16* Bt, int ldb, int n0, int k0, int lane) {
  int col = lane & 15;
  int kb  = k0 + ((lane >> 4) << 4);
  const _Float16* p = Bt + (size_t)(n0 + col) * ldb + kb;
  U16H u;
  u.q[0] = *(const uint4*)(p);
  u.q[1] = *(const uint4*)(p + 8);
  return u.v;
}

// ---------------- LayerNorm -> f16 activations ----------------
__global__ void ln_kernel(const float* __restrict__ z, const float* __restrict__ sc,
                          const float* __restrict__ bi, _Float16* __restrict__ zn) {
  int row = blockIdx.x;
  int tid = threadIdx.x;
  __shared__ float rs[256], rq[256];
  const float* zr = z + (size_t)row * D_;
  float vals[4];
  float s = 0.f, q = 0.f;
#pragma unroll
  for (int i = 0; i < 4; ++i) {
    float x = zr[tid + i * 256];
    vals[i] = x; s += x; q += x * x;
  }
  rs[tid] = s; rq[tid] = q;
  __syncthreads();
  for (int off = 128; off > 0; off >>= 1) {
    if (tid < off) { rs[tid] += rs[tid + off]; rq[tid] += rq[tid + off]; }
    __syncthreads();
  }
  float mu   = rs[0] * (1.0f / D_);
  float var  = rq[0] * (1.0f / D_) - mu * mu;
  float rstd = rsqrtf(var + 1e-5f);
  _Float16* o = zn + (size_t)row * D_;
#pragma unroll
  for (int i = 0; i < 4; ++i) {
    int c = tid + i * 256;
    o[c] = (_Float16)((vals[i] - mu) * rstd * sc[c] + bi[c]);
  }
}

// ---------------- Weight packing (fp32 -> f16, transposed / de-interleaved) ----------------
// Packed QKV weight order: n' = which*1024 + h*64 + hd ; Wt[n'][k] = w_qkv[k][(h*64+hd)*3+which]
__global__ void pack_wqkv(const float* __restrict__ w, _Float16* __restrict__ wt) {
  int idx   = blockIdx.x * 256 + threadIdx.x;     // 3072*1024 elements
  int np    = idx >> 10;
  int k     = idx & 1023;
  int which = np >> 10;
  int rem   = np & 1023;
  int h     = rem >> 6;
  int hd    = rem & 63;
  int f     = (h * HD_ + hd) * 3 + which;
  wt[idx] = (_Float16)w[(size_t)k * (3 * D_) + f];
}

__global__ void pack_wproj(const float* __restrict__ w, _Float16* __restrict__ wt) {
  int idx = blockIdx.x * 256 + threadIdx.x;       // 1024*1024 elements
  int n = idx >> 10, k = idx & 1023;
  wt[idx] = (_Float16)w[(size_t)k * D_ + n];
}

// ---------------- QKV GEMM: [4096,1024] x [1024,3072], 2x4 register-blocked ----------------
// One wave computes a 32x64 C block: 2 A-frags x 4 B-frags -> 8 WMMA / k-step,
// ~21 flops per operand byte. Epilogue scatters Q,K row-major per head and V
// transposed ([b][h][hd][n]) so the P*V B-operand is contiguous.
__global__ void __launch_bounds__(256) qkv_gemm(const _Float16* __restrict__ zn,
                                                const _Float16* __restrict__ wt,
                                                _Float16* __restrict__ Qb,
                                                _Float16* __restrict__ Kb,
                                                _Float16* __restrict__ Vt) {
  int lane = threadIdx.x & 31;
  int w    = blockIdx.x * 8 + (threadIdx.x >> 5);
  int tn   = w % 48;                 // 3072 / 64 column strips
  int tm   = w / 48;                 // 4096 / 32 row blocks
  int m0   = tm * 32, n0 = tn * 64;

  v8f acc[2][4];
#pragma unroll
  for (int i = 0; i < 2; ++i)
#pragma unroll
    for (int j = 0; j < 4; ++j) acc[i][j] = (v8f){};

  for (int k0 = 0; k0 < 1024; k0 += 32) {
    __builtin_prefetch(wt + (size_t)n0 * 1024 + k0 + 256, 0, 0);
    v16h a0 = load_a16(zn, 1024, m0,      k0, lane);
    v16h a1 = load_a16(zn, 1024, m0 + 16, k0, lane);
    v16h b0 = load_bt16(wt, 1024, n0,      k0, lane);
    v16h b1 = load_bt16(wt, 1024, n0 + 16, k0, lane);
    v16h b2 = load_bt16(wt, 1024, n0 + 32, k0, lane);
    v16h b3 = load_bt16(wt, 1024, n0 + 48, k0, lane);
    acc[0][0] = __builtin_amdgcn_wmma_f32_16x16x32_f16(false, a0, false, b0, (short)0, acc[0][0], false, false);
    acc[1][0] = __builtin_amdgcn_wmma_f32_16x16x32_f16(false, a1, false, b0, (short)0, acc[1][0], false, false);
    acc[0][1] = __builtin_amdgcn_wmma_f32_16x16x32_f16(false, a0, false, b1, (short)0, acc[0][1], false, false);
    acc[1][1] = __builtin_amdgcn_wmma_f32_16x16x32_f16(false, a1, false, b1, (short)0, acc[1][1], false, false);
    acc[0][2] = __builtin_amdgcn_wmma_f32_16x16x32_f16(false, a0, false, b2, (short)0, acc[0][2], false, false);
    acc[1][2] = __builtin_amdgcn_wmma_f32_16x16x32_f16(false, a1, false, b2, (short)0, acc[1][2], false, false);
    acc[0][3] = __builtin_amdgcn_wmma_f32_16x16x32_f16(false, a0, false, b3, (short)0, acc[0][3], false, false);
    acc[1][3] = __builtin_amdgcn_wmma_f32_16x16x32_f16(false, a1, false, b3, (short)0, acc[1][3], false, false);
  }

  int half = lane >> 4, cn = lane & 15;
  // A 64-wide column strip stays inside one (which, h) group: 64 | 1024.
  int which = n0 >> 10;
  int h     = (n0 & 1023) >> 6;
#pragma unroll
  for (int j = 0; j < 4; ++j) {
    int hd = j * 16 + cn;
#pragma unroll
    for (int i = 0; i < 2; ++i) {
#pragma unroll
      for (int r = 0; r < 8; ++r) {
        int t  = m0 + i * 16 + r + half * 8;
        int bb = t >> 11;
        int n  = t & 2047;
        _Float16 v = (_Float16)acc[i][j][r];
        if (which == 0)      Qb[(((size_t)(bb * H_ + h)) * N_ + n) * HD_ + hd] = v;
        else if (which == 1) Kb[(((size_t)(bb * H_ + h)) * N_ + n) * HD_ + hd] = v;
        else                 Vt[(((size_t)(bb * H_ + h)) * HD_ + hd) * N_ + n] = v;
      }
    }
  }
}

// ---------------- Flash-style attention: one block per (b,h,16-query block) ----------------
__global__ void __launch_bounds__(128) attn_kernel(const _Float16* __restrict__ Qb,
                                                   const _Float16* __restrict__ Kb,
                                                   const _Float16* __restrict__ Vt,
                                                   _Float16* __restrict__ Ob) {
  __shared__ float Ssc[16 * CHUNK];                       // 32 KB score buffer
  __shared__ __attribute__((aligned(16))) _Float16 Pld[16 * CHUNK];  // 16 KB P (A-operand layout source)
  __shared__ float redm[16][8], reds[16][8];
  __shared__ float rowm[16], rowc[16], rowsum[16];

  int tid  = threadIdx.x;
  int lane = tid & 31;
  int wv   = tid >> 5;                                    // 4 waves
  int mb = blockIdx.x & 127;
  int h  = (blockIdx.x >> 7) & 15;
  int b  = blockIdx.x >> 11;
  int m0 = mb * 16;

  const _Float16* Qh = Qb + ((size_t)(b * H_ + h)) * N_ * HD_;
  const _Float16* Kh = Kb + ((size_t)(b * H_ + h)) * N_ * HD_;
  const _Float16* Vh = Vt + ((size_t)(b * H_ + h)) * HD_ * N_;

  v16h aq0 = load_a16(Qh, HD_, m0, 0, lane);
  v16h aq1 = load_a16(Qh, HD_, m0, 32, lane);
  v8f  o   = {};

  if (tid < 16) { rowm[tid] = -3.0e38f; rowsum[tid] = 0.f; }
  __syncthreads();

  int half = lane >> 4, cn = lane & 15;
  int row8 = tid >> 3, sub8 = tid & 7;

  for (int c0 = 0; c0 < N_; c0 += CHUNK) {
    // ---- S = (Q K^T)/8 tiles -> LDS. K rows are B-columns (contiguous). ----
#pragma unroll
    for (int i = 0; i < 8; ++i) {
      int tt = wv * 8 + i;
      int nk = c0 + tt * 16;
      v8f s = {};
      s = __builtin_amdgcn_wmma_f32_16x16x32_f16(false, aq0, false,
            load_bt16(Kh, HD_, nk, 0, lane),  (short)0, s, false, false);
      s = __builtin_amdgcn_wmma_f32_16x16x32_f16(false, aq1, false,
            load_bt16(Kh, HD_, nk, 32, lane), (short)0, s, false, false);
      int colr = tt * 16 + cn;
#pragma unroll
      for (int j = 0; j < 8; ++j)
        Ssc[(j + half * 8) * CHUNK + colr] = s[j] * 0.125f;
    }
    __syncthreads();

    // ---- chunk row max ----
    {
      const float* sr = Ssc + row8 * CHUNK + sub8 * 64;
      float mx = -3.0e38f;
      for (int i2 = 0; i2 < 64; ++i2) mx = fmaxf(mx, sr[i2]);
      redm[row8][sub8] = mx;
    }
    __syncthreads();
    if (tid < 16) {
      float mx = redm[tid][0];
      for (int i2 = 1; i2 < 8; ++i2) mx = fmaxf(mx, redm[tid][i2]);
      float mnew = fmaxf(rowm[tid], mx);
      rowc[tid] = __expf(rowm[tid] - mnew);
      rowm[tid] = mnew;
    }
    __syncthreads();

    // ---- P = exp(S - m), f16 into LDS; chunk row sum ----
    {
      float mn = rowm[row8];
      const float* sr = Ssc + row8 * CHUNK + sub8 * 64;
      _Float16*   pr = Pld + row8 * CHUNK + sub8 * 64;
      float sm = 0.f;
      for (int i2 = 0; i2 < 64; ++i2) {
        float p = __expf(sr[i2] - mn);
        pr[i2] = (_Float16)p;
        sm += p;
      }
      reds[row8][sub8] = sm;
    }
    __syncthreads();
    if (tid < 16) {
      float sm = 0.f;
      for (int i2 = 0; i2 < 8; ++i2) sm += reds[tid][i2];
      rowsum[tid] = rowsum[tid] * rowc[tid] + sm;
    }
    __syncthreads();

    // ---- rescale O, then O += P V (per-wave 16-wide hd column tile) ----
#pragma unroll
    for (int j = 0; j < 8; ++j) o[j] *= rowc[j + half * 8];
#pragma unroll 4
    for (int kk = 0; kk < CHUNK; kk += 32) {
      v16h pa = load_a16((const _Float16*)Pld, CHUNK, 0, kk, lane);
      v16h vb = load_bt16(Vh, N_, wv * 16, c0 + kk, lane);
      o = __builtin_amdgcn_wmma_f32_16x16x32_f16(false, pa, false, vb, (short)0, o, false, false);
    }
    __syncthreads();
  }

  // ---- normalize and store merged-head O (f16, A-operand for proj GEMM) ----
#pragma unroll
  for (int j = 0; j < 8; ++j) {
    int r   = j + half * 8;
    int tok = b * N_ + m0 + r;
    float val = o[j] / rowsum[r];
    Ob[(size_t)tok * D_ + h * HD_ + wv * 16 + cn] = (_Float16)val;
  }
}

// ---------------- Output projection + bias + residual (fp32 out), 2x4 blocked ----------------
__global__ void __launch_bounds__(256) proj_gemm(const _Float16* __restrict__ Ob,
                                                 const _Float16* __restrict__ wt,
                                                 const float* __restrict__ z,
                                                 const float* __restrict__ bp,
                                                 float* __restrict__ out) {
  int lane = threadIdx.x & 31;
  int w    = blockIdx.x * 8 + (threadIdx.x >> 5);
  int tn   = w & 15;                 // 1024 / 64 column strips
  int tm   = w >> 4;                 // 4096 / 32 row blocks
  int m0   = tm * 32, n0 = tn * 64;

  v8f acc[2][4];
#pragma unroll
  for (int i = 0; i < 2; ++i)
#pragma unroll
    for (int j = 0; j < 4; ++j) acc[i][j] = (v8f){};

  for (int k0 = 0; k0 < 1024; k0 += 32) {
    __builtin_prefetch(wt + (size_t)n0 * 1024 + k0 + 256, 0, 0);
    v16h a0 = load_a16(Ob, 1024, m0,      k0, lane);
    v16h a1 = load_a16(Ob, 1024, m0 + 16, k0, lane);
    v16h b0 = load_bt16(wt, 1024, n0,      k0, lane);
    v16h b1 = load_bt16(wt, 1024, n0 + 16, k0, lane);
    v16h b2 = load_bt16(wt, 1024, n0 + 32, k0, lane);
    v16h b3 = load_bt16(wt, 1024, n0 + 48, k0, lane);
    acc[0][0] = __builtin_amdgcn_wmma_f32_16x16x32_f16(false, a0, false, b0, (short)0, acc[0][0], false, false);
    acc[1][0] = __builtin_amdgcn_wmma_f32_16x16x32_f16(false, a1, false, b0, (short)0, acc[1][0], false, false);
    acc[0][1] = __builtin_amdgcn_wmma_f32_16x16x32_f16(false, a0, false, b1, (short)0, acc[0][1], false, false);
    acc[1][1] = __builtin_amdgcn_wmma_f32_16x16x32_f16(false, a1, false, b1, (short)0, acc[1][1], false, false);
    acc[0][2] = __builtin_amdgcn_wmma_f32_16x16x32_f16(false, a0, false, b2, (short)0, acc[0][2], false, false);
    acc[1][2] = __builtin_amdgcn_wmma_f32_16x16x32_f16(false, a1, false, b2, (short)0, acc[1][2], false, false);
    acc[0][3] = __builtin_amdgcn_wmma_f32_16x16x32_f16(false, a0, false, b3, (short)0, acc[0][3], false, false);
    acc[1][3] = __builtin_amdgcn_wmma_f32_16x16x32_f16(false, a1, false, b3, (short)0, acc[1][3], false, false);
  }

  int half = lane >> 4, cn = lane & 15;
#pragma unroll
  for (int j = 0; j < 4; ++j) {
    int col  = n0 + j * 16 + cn;
    float bias = bp[col];
#pragma unroll
    for (int i = 0; i < 2; ++i) {
#pragma unroll
      for (int r = 0; r < 8; ++r) {
        int t = m0 + i * 16 + r + half * 8;
        out[(size_t)t * D_ + col] = acc[i][j][r] + bias + z[(size_t)t * D_ + col];
      }
    }
  }
}

extern "C" void kernel_launch(void* const* d_in, const int* in_sizes, int n_in,
                              void* d_out, int out_size, void* d_ws, size_t ws_size,
                              hipStream_t stream) {
  const float* z        = (const float*)d_in[0];
  const float* ln_scale = (const float*)d_in[1];
  const float* ln_bias  = (const float*)d_in[2];
  const float* w_qkv    = (const float*)d_in[3];
  const float* w_proj   = (const float*)d_in[4];
  const float* b_proj   = (const float*)d_in[5];
  float* out = (float*)d_out;

  // Workspace layout (f16 elements), total 48 MiB.
  _Float16* ws = (_Float16*)d_ws;
  size_t off = 0;
  _Float16* zn     = ws + off; off += (size_t)NT_ * D_;          // 8 MiB
  _Float16* wqkvt  = ws + off; off += (size_t)3 * D_ * D_;       // 6 MiB
  _Float16* wprojt = ws + off; off += (size_t)D_ * D_;           // 2 MiB
  _Float16* Qb     = ws + off; off += (size_t)B_ * H_ * N_ * HD_;// 8 MiB
  _Float16* Kb     = ws + off; off += (size_t)B_ * H_ * N_ * HD_;// 8 MiB
  _Float16* Vt     = ws + off; off += (size_t)B_ * H_ * N_ * HD_;// 8 MiB
  _Float16* Ob     = ws + off;                                   // 8 MiB

  pack_wqkv <<<(3 * D_ * D_) / 256, 256, 0, stream>>>(w_qkv, wqkvt);
  pack_wproj<<<(D_ * D_) / 256,     256, 0, stream>>>(w_proj, wprojt);
  ln_kernel <<<NT_, 256, 0, stream>>>(z, ln_scale, ln_bias, zn);

  // 128 x 48 (32x64) blocks, 8 waves/block -> 768 blocks
  qkv_gemm<<<(128 * 48) / 8, 256, 0, stream>>>(zn, wqkvt, Qb, Kb, Vt);

  // one block per (b, h, 16-query block)
  attn_kernel<<<B_ * H_ * (N_ / 16), 128, 0, stream>>>(Qb, Kb, Vt, Ob);

  // 128 x 16 (32x64) blocks, 8 waves/block -> 256 blocks
  proj_gemm<<<(128 * 16) / 8, 256, 0, stream>>>(Ob, wprojt, z, b_proj, out);
}
